// DistributedTPForward_64836826300874
// MI455X (gfx1250) — matrix operand order repaired and checked by reading, not compiled, not run on hardware
//
#include <hip/hip_runtime.h>

// ---------------------------------------------------------------------------
// Fused tensor-parallel MLP forward for MI455X (gfx1250, wave32, WMMA + TDM).
//   out = relu(X @ W1^T + b1) @ W2' + b2
//   X:[32768,1024] f32, W1:[4096,1024], W2:[4,1024,1024] ([W,D,HS]), out:[32768,1024]
//
// Memory-bound (~300 MB mandatory HBM traffic @ 23.3 TB/s ~= 13 us):
//   * pre-convert X/W1/W2 to bf16 into d_ws; weights (16 MB) + X (64 MB) are L2-resident
//   * fuse both GEMMs; hidden tiles live only in LDS (saves ~1 GB HBM r+w)
//   * X tile staged to LDS once per block via TENSOR_LOAD_TO_LDS (TDM, TENSORcnt),
//     with D#-generated padding so stage-1 A fragments are 16B-aligned ds_load_b128
//   * all math on v_wmma_f32_16x16x32_bf16 with fp32 accumulation
// ---------------------------------------------------------------------------

#define N_TOK   32768
#define DMODEL  1024
#define HTOT    4096              // W * HS = 4 * 1024
#define HSH     1024

#define TOK_PER_BLK 32            // token tile per block (2 WMMA M-strips)
#define CHUNK       512           // hidden columns staged in LDS per iteration
#define LDS_STRIDE  520           // hidden: CHUNK + 8 halfword pad (conflict-free b128)

// X tile LDS layout produced by the TDM descriptor:
//   pad 12 dwords every 256 dwords  ->  half-row chunk = 512 bf16 + 24 pad = 536 hw
//   element (r,k) at halfword index (2r + (k>>9))*536 + (k & 511)
#define XCHUNK_HW   536
#define XTILE_HW    (TOK_PER_BLK * 2 * XCHUNK_HW)          // 34304 halfwords
#define XTILE_BYTES (XTILE_HW * 2)                          // 68608 bytes
#define HID_OFF_HW  XTILE_HW
#define LDS_BYTES   (XTILE_BYTES + TOK_PER_BLK * LDS_STRIDE * 2)   // 101888 B

typedef __attribute__((ext_vector_type(16))) __bf16       v16bf;
typedef __attribute__((ext_vector_type(8)))  float        v8f;
typedef __attribute__((ext_vector_type(8)))  unsigned int v8u;
typedef __attribute__((ext_vector_type(4)))  unsigned int v4u;
typedef __attribute__((ext_vector_type(8)))  int          v8i;
typedef __attribute__((ext_vector_type(4)))  int          v4i;

static __device__ __forceinline__ unsigned short f2bf(float f) {
  unsigned int u = __builtin_bit_cast(unsigned int, f);
  u += 0x7fffu + ((u >> 16) & 1u);            // round-to-nearest-even
  return (unsigned short)(u >> 16);
}

// 16 bf16 per lane as two 16B loads (global or LDS), per ISA fragment layouts.
static __device__ __forceinline__ v16bf load_frag(const unsigned short* p, int off2) {
  uint4 q0 = *(const uint4*)(p);
  uint4 q1 = *(const uint4*)(p + off2);
  v8u t = {q0.x, q0.y, q0.z, q0.w, q1.x, q1.y, q1.z, q1.w};
  return __builtin_bit_cast(v16bf, t);
}

__global__ void cvt_f32_to_bf16(const float* __restrict__ s,
                                unsigned short* __restrict__ d, int n) {
  int i = (blockIdx.x * blockDim.x + threadIdx.x) * 4;
  if (i < n) {
    float4 f = *(const float4*)(s + i);
    ushort4 o;
    o.x = f2bf(f.x); o.y = f2bf(f.y); o.z = f2bf(f.z); o.w = f2bf(f.w);
    *(ushort4*)(d + i) = o;
  }
}

__global__ __launch_bounds__(256) void fused_tp_mlp(
    const unsigned short* __restrict__ xb,   // [N_TOK, DMODEL] bf16
    const unsigned short* __restrict__ w1b,  // [HTOT, DMODEL]  bf16 (K-contig)
    const unsigned short* __restrict__ w2b,  // [W, DMODEL, HSH] bf16 (K=h contig)
    const float* __restrict__ b1,            // [HTOT]
    const float* __restrict__ b2,            // [DMODEL]
    float* __restrict__ out) {               // [N_TOK, DMODEL]
  extern __shared__ unsigned short smem[];
  unsigned short* xlds = smem;               // TDM-padded X tile
  unsigned short* hid  = smem + HID_OFF_HW;  // [TOK_PER_BLK][LDS_STRIDE]

  const int tid  = threadIdx.x;
  const int wv   = tid >> 5;                 // wave32: 8 waves / block
  const int lane = tid & 31;
  const int n16  = lane & 15;                // row (A) / col (B,C) index
  const int half = lane >> 4;                // K-half selector per ISA layout
  const int tok0 = blockIdx.x * TOK_PER_BLK;

  // ---- Stage 0: DMA the 32x1024 bf16 X tile into LDS via the TDM ----------
#if __has_builtin(__builtin_amdgcn_tensor_load_to_lds) && \
    __has_builtin(__builtin_amdgcn_s_wait_tensorcnt)
  if (wv == 0) {
    const unsigned long long ga =
        (unsigned long long)(const void*)(xb + (size_t)tok0 * DMODEL);
    const unsigned lds_off = (unsigned)(unsigned long long)(const void*)xlds;
    v4u g0 = {
        1u,                                        // count=1 (valid descriptor)
        lds_off,                                   // lds_addr (bytes)
        (unsigned)ga,                              // global_addr[31:0]
        (unsigned)((ga >> 32) & 0x01FFFFFFu) | (2u << 30)  // addr[56:32] | type=2
    };
    v8i g1 = {
        (int)((1u << 16) |                         // data_size = 2 bytes
              (1u << 20) |                         // pad_enable
              (7u << 22) |                         // pad_interval = 256 dwords
              (11u << 25)),                        // pad_amount  = 12 dwords
        (int)(1024u << 16),                        // tensor_dim0 = 1024 (lo16)
        (int)(32u << 16),                          // tensor_dim0 hi=0 | tensor_dim1=32
        (int)(1024u << 16),                        // tensor_dim1 hi=0 | tile_dim0=1024
        (int)32,                                   // tile_dim1 = 32, tile_dim2 = 0 (2D)
        (int)1024,                                 // tensor_dim0_stride = 1024 (lo32)
        0, 0
    };
    v4i gz4 = {0, 0, 0, 0};
    v8i gz8 = {0, 0, 0, 0, 0, 0, 0, 0};
    // clang-23 / therock-10.0 form: 6 args (extra int32x8 group, zero-filled).
    __builtin_amdgcn_tensor_load_to_lds(g0, g1, gz4, gz4, gz8, 0);
    __builtin_amdgcn_s_wait_tensorcnt(0);
  }
#else
  // Cooperative fallback with the identical padded layout.
  for (int idx = tid; idx < TOK_PER_BLK * DMODEL / 8; idx += 256) {
    const int e = idx * 8, r = e >> 10, k = e & (DMODEL - 1);
    const uint4 q = *(const uint4*)(xb + (size_t)(tok0 + r) * DMODEL + k);
    *(uint4*)(xlds + (2 * r + (k >> 9)) * XCHUNK_HW + (k & 511)) = q;
  }
#endif
  __syncthreads();

  // Persistent output accumulators: wave owns 32 tokens x 128 d-cols.
  v8f c2[2][8];
  const v8f vzero = {0.f, 0.f, 0.f, 0.f, 0.f, 0.f, 0.f, 0.f};
#pragma unroll
  for (int i = 0; i < 2; ++i)
#pragma unroll
    for (int j = 0; j < 8; ++j) c2[i][j] = vzero;

  for (int c = 0; c < HTOT / CHUNK; ++c) {
    const int hbase = c * CHUNK;

    // ---- Stage 1: hidden chunk = relu(X * W1^T + b1), wave -> 32x64 slab ----
    v8f c1[2][4];
#pragma unroll
    for (int i = 0; i < 2; ++i)
#pragma unroll
      for (int j = 0; j < 4; ++j) c1[i][j] = vzero;

    const int hb = hbase + wv * 64;
    for (int kk = 0; kk < DMODEL; kk += 32) {
      v16bf a[2], b[4];
#pragma unroll
      for (int ms = 0; ms < 2; ++ms) {       // A 16x32 from LDS (TDM layout)
        const unsigned short* p =
            xlds + (2 * (ms * 16 + n16) + (kk >> 9)) * XCHUNK_HW +
            (kk & 511) + half * 8;
        a[ms] = load_frag(p, 16);
      }
#pragma unroll
      for (int ns = 0; ns < 4; ++ns) {       // B 32x16: lane col = n16, 16 K's
        const unsigned short* p =
            w1b + (size_t)(hb + ns * 16 + n16) * DMODEL + kk + half * 8;
        b[ns] = load_frag(p, 16);
      }
#pragma unroll
      for (int ms = 0; ms < 2; ++ms)
#pragma unroll
        for (int ns = 0; ns < 4; ++ns)
          c1[ms][ns] = __builtin_amdgcn_wmma_f32_16x16x32_bf16(
              false, a[ms], false, b[ns], (short)0, c1[ms][ns], false, false);
    }

    __syncthreads();  // previous chunk's stage-2 readers done before we rewrite

    // bias + ReLU + bf16, scatter into LDS row-major (padded stride).
#pragma unroll
    for (int ms = 0; ms < 2; ++ms)
#pragma unroll
      for (int ns = 0; ns < 4; ++ns) {
        const int col  = wv * 64 + ns * 16 + n16;
        const float bv = b1[hbase + col];
#pragma unroll
        for (int i = 0; i < 8; ++i) {        // C v[i]: M = half*8 + i, N = n16
          float v = c1[ms][ns][i] + bv;
          v = v > 0.f ? v : 0.f;
          hid[(ms * 16 + half * 8 + i) * LDS_STRIDE + col] = f2bf(v);
        }
      }
    __syncthreads();

    // ---- Stage 2: c2 += hidden(LDS) * W2 slab, wave -> its 128 d-cols ----
    const int db = wv * 128;
    for (int kk = 0; kk < CHUNK; kk += 32) {
      v16bf a[2], b[8];
#pragma unroll
      for (int ms = 0; ms < 2; ++ms) {
        const unsigned short* p =
            hid + (ms * 16 + n16) * LDS_STRIDE + kk + half * 8;
        a[ms] = load_frag(p, 16);            // ds_load_b128 x2, conflict-free
      }
      const int hg  = hbase + kk + half * 8;   // global hidden index
      const int wsd = hg >> 10;                // shard w
      const int hl  = hg & (HSH - 1);          // h within shard
#pragma unroll
      for (int ns = 0; ns < 8; ++ns) {
        const unsigned short* p =
            w2b + (size_t)(wsd * DMODEL + db + ns * 16 + n16) * HSH + hl;
        b[ns] = load_frag(p, 16);
      }
#pragma unroll
      for (int ms = 0; ms < 2; ++ms)
#pragma unroll
        for (int ns = 0; ns < 8; ++ns)
          c2[ms][ns] = __builtin_amdgcn_wmma_f32_16x16x32_bf16(
              false, a[ms], false, b[ns], (short)0, c2[ms][ns], false, false);
    }
    // next iteration's pre-write barrier orders these reads vs. rewrites
  }

  // ---- Epilogue: out = c2 + b2 ----
#pragma unroll
  for (int ms = 0; ms < 2; ++ms)
#pragma unroll
    for (int ns = 0; ns < 8; ++ns) {
      const int d    = wv * 128 + ns * 16 + n16;
      const float bv = b2[d];
#pragma unroll
      for (int i = 0; i < 8; ++i) {
        const int row = tok0 + ms * 16 + half * 8 + i;
        out[(size_t)row * DMODEL + d] = c2[ms][ns][i] + bv;
      }
    }
}

extern "C" void kernel_launch(void* const* d_in, const int* in_sizes, int n_in,
                              void* d_out, int out_size, void* d_ws, size_t ws_size,
                              hipStream_t stream) {
  const float* x  = (const float*)d_in[0];  // [N, D]
  const float* w1 = (const float*)d_in[1];  // [W, HS, D] == [HTOT, D]
  const float* b1 = (const float*)d_in[2];  // [W, HS]    == [HTOT]
  const float* w2 = (const float*)d_in[3];  // [W, D, HS]
  const float* b2 = (const float*)d_in[4];  // [D]
  float* out = (float*)d_out;

  // d_ws layout: bf16 copies of X (64 MB), W1 (8 MB), W2 (8 MB) = 80 MB.
  unsigned short* xb  = (unsigned short*)d_ws;
  unsigned short* w1b = xb + (size_t)N_TOK * DMODEL;
  unsigned short* w2b = w1b + (size_t)HTOT * DMODEL;

  const int nx = N_TOK * DMODEL;  // 33,554,432
  const int nw = HTOT * DMODEL;   //  4,194,304
  cvt_f32_to_bf16<<<nx / 1024, 256, 0, stream>>>(x, xb, nx);
  cvt_f32_to_bf16<<<nw / 1024, 256, 0, stream>>>(w1, w1b, nw);
  cvt_f32_to_bf16<<<nw / 1024, 256, 0, stream>>>(w2, w2b, nw);

  fused_tp_mlp<<<N_TOK / TOK_PER_BLK, 256, LDS_BYTES, stream>>>(
      xb, w1b, w2b, b1, b2, out);
}
